// HGCN_63110249447953
// MI455X (gfx1250) — compile-verified, matrix operand order checked by reading, not run on hardware
//
#include <hip/hip_runtime.h>
#include <cstdint>

// ---------------------------------------------------------------------------
// Types for CDNA5 WMMA (wave32, 16x16x32 bf16 -> f32)
// ---------------------------------------------------------------------------
typedef __bf16 bf16_t;
typedef __attribute__((ext_vector_type(8)))  __bf16 bf16x8;
typedef __attribute__((ext_vector_type(16))) __bf16 bf16x16;
typedef __attribute__((ext_vector_type(8)))  float  f32x8;

#if defined(__has_builtin)
#if __has_builtin(__builtin_amdgcn_sched_group_barrier)
#define SCHED_GROUP(mask, size) __builtin_amdgcn_sched_group_barrier((mask), (size), 0)
#endif
#endif
#ifndef SCHED_GROUP
#define SCHED_GROUP(mask, size)
#endif

// ---------------------------------------------------------------------------
// Problem dimensions
// ---------------------------------------------------------------------------
#define B_ 8
#define F_ 321
#define T_ 1000
#define H_ 4200

constexpr int FPAD = 352;   // F padded to multiple of 32 (11 k-steps)
constexpr int HPAD = 4224;  // H padded to 264 h-tiles (33 per wave, 8 waves)
constexpr int NHT  = HPAD / 16;          // 264
constexpr int TPW  = NHT / 8;            // 33 h-tiles per wave
constexpr int TP2  = 1008;  // T padded to multiple of 16 for GEMM B-matrix
constexpr int NTT  = 63;    // time tiles of 16 (63*16 = 1008 >= 1000)
constexpr int FP   = 325;   // F + 4 (freq pad 2 each side)
constexpr int TP   = 1012;  // 1 (left causal pad) + 1008 + slack
constexpr int CP   = 16;    // channels per pixel in NHWC-padded layout

// Workspace layout (bytes, all 256-aligned)
constexpr size_t OFF_MAGT = 0;
constexpr size_t SZ_MAGT  = (size_t)B_ * TP2 * FPAD * 2;     // bf16 magT [B][TP2][FPAD]
constexpr size_t OFF_MBF  = OFF_MAGT + SZ_MAGT;
constexpr size_t SZ_MBF   = (size_t)HPAD * FPAD * 2;         // bf16 M    [HPAD][FPAD]
constexpr size_t OFF_POS  = OFF_MBF + SZ_MBF;
constexpr size_t SZ_POS   = 32256;                           // int pos   [B][T]
constexpr size_t OFF_GATE = OFF_POS + SZ_POS;
constexpr size_t SZ_GATE  = (size_t)B_ * F_ * T_ * 4;        // f32 gate  [B][F][T]
constexpr size_t OFF_WPK  = OFF_GATE + SZ_GATE;
constexpr size_t SZ_WPK   = (size_t)12 * 10 * 16 * 32 * 2;   // 12 packed conv weights
constexpr size_t OFF_BUF0 = OFF_WPK + SZ_WPK;
constexpr size_t SZ_BUF   = (size_t)B_ * FP * TP * CP * 2 + 256; // +slack for frag overrun
constexpr size_t OFF_BUF1 = OFF_BUF0 + SZ_BUF;
constexpr size_t OFF_BUF2 = OFF_BUF1 + SZ_BUF;
constexpr size_t WS_TOTAL = OFF_BUF2 + SZ_BUF;

// ---------------------------------------------------------------------------
// Fragment helpers. Doc layout (CDNA5 ISA): lane L, hi = L>>4
// holds, for its row M=L%16 (A) / col N=L%16 (B), K values {hi*8..hi*8+7}
// and {16+hi*8..16+hi*8+7} -> two contiguous 16-byte chunks.
// ---------------------------------------------------------------------------
__device__ __forceinline__ bf16x16 cat8(bf16x8 lo, bf16x8 hh) {
  return __builtin_shufflevector(lo, hh, 0, 1, 2, 3, 4, 5, 6, 7,
                                         8, 9, 10, 11, 12, 13, 14, 15);
}
__device__ __forceinline__ bf16x16 load_frag(const bf16_t* __restrict__ base, int hi) {
  bf16x8 lo = *(const bf16x8*)(base + hi * 8);
  bf16x8 hh = *(const bf16x8*)(base + 16 + hi * 8);
  return cat8(lo, hh);
}

// ---------------------------------------------------------------------------
// Utility kernels
// ---------------------------------------------------------------------------
__global__ void zero_kernel(uint32_t* __restrict__ p, size_t nwords) {
  size_t i = (size_t)blockIdx.x * blockDim.x + threadIdx.x;
  size_t stride = (size_t)gridDim.x * blockDim.x;
  for (; i < nwords; i += stride) p[i] = 0u;
}

// mag [B][F][T] f32  ->  magT [B][TP2][FPAD] bf16 (f contiguous per t)
__global__ void cvt_magT_kernel(const float* __restrict__ mag, bf16_t* __restrict__ magT) {
  int idx = blockIdx.x * blockDim.x + threadIdx.x;
  if (idx >= B_ * F_ * T_) return;
  int t = idx % T_;
  int f = (idx / T_) % F_;
  int b = idx / (F_ * T_);
  magT[((size_t)b * TP2 + t) * FPAD + f] = (bf16_t)mag[((size_t)b * F_ + f) * T_ + t];
}

// M [H][F] f32 -> Mbf [HPAD][FPAD] bf16 (zero padded by the ws zero pass)
__global__ void cvt_M_kernel(const float* __restrict__ M, bf16_t* __restrict__ Mbf) {
  int idx = blockIdx.x * blockDim.x + threadIdx.x;
  if (idx >= H_ * F_) return;
  int f = idx % F_;
  int h = idx / F_;
  Mbf[(size_t)h * FPAD + f] = (bf16_t)M[(size_t)h * F_ + f];
}

// in_feature -> channel 0 of NHWC-padded buf0 (rest already zero)
__global__ void init_act_kernel(const float* __restrict__ x, bf16_t* __restrict__ buf) {
  int idx = blockIdx.x * blockDim.x + threadIdx.x;
  if (idx >= B_ * F_ * T_) return;
  int t = idx % T_;
  int f = (idx / T_) % F_;
  int b = idx / (F_ * T_);
  buf[(((size_t)b * FP + (f + 2)) * TP + (t + 1)) * CP + 0] = (bf16_t)x[idx];
}

// conv weight [cout][cin][5][2] f32 -> packed bf16 [10 taps][16 co][32 ci]
__global__ void pack_w_kernel(const float* __restrict__ w, bf16_t* __restrict__ wp,
                              int cout, int cin) {
  int i = blockIdx.x * blockDim.x + threadIdx.x;
  if (i >= 10 * 16 * 32) return;
  int tap = i >> 9;
  int co  = (i >> 5) & 15;
  int ci  = i & 31;
  int kh = tap >> 1, kw = tap & 1;
  float v = 0.0f;
  if (co < cout && ci < cin) v = w[(((size_t)co * cin + ci) * 5 + kh) * 2 + kw];
  wp[i] = (bf16_t)v;
}

// ---------------------------------------------------------------------------
// Harmonic integral: bf16 WMMA GEMM (nominee) fused with per-column argmax.
// Grid: (63 t-tiles, 8 batch); block 32x8 (8 waves). Each wave owns 33
// h-tiles (stride 8). sched_group_barrier forces 23 clustered VMEM reads
// (22 A-frag loads + prefetch) ahead of an 11-wmma burst; launch_bounds
// (min 1 wave/EU) relaxes the allocator's occupancy-driven VGPR cap so the
// ~200 live VGPRs (88 B-frags + 88 A-tile + acc) stay resident, no spills.
// ---------------------------------------------------------------------------
__global__ void __launch_bounds__(256, 1)
gemm_argmax_kernel(const bf16_t* __restrict__ magT,
                   const bf16_t* __restrict__ Mbf,
                   int* __restrict__ pos) {
  const int tt   = blockIdx.x;
  const int b    = blockIdx.y;
  const int lane = threadIdx.x;      // 0..31
  const int wv   = threadIdx.y;      // 0..7
  const int n  = lane & 15;
  const int hi = lane >> 4;
  const int t  = tt * 16 + n;

  // Preload all 11 B fragments for this wave's 16 time columns.
  const bf16_t* brow = magT + ((size_t)b * TP2 + t) * FPAD;
  bf16x16 bfrag[11];
#pragma unroll
  for (int kk = 0; kk < 11; ++kk) bfrag[kk] = load_frag(brow + kk * 32, hi);

  float bestv = -3.0e38f;
  int   besth = 0;

#pragma unroll 1
  for (int i = 0; i < TPW; ++i) {
    const int h0 = (wv + i * 8) * 16;
    const bf16_t* arow = Mbf + (size_t)(h0 + n) * FPAD;
    __builtin_prefetch(Mbf + (size_t)(h0 + 128 + n) * FPAD, 0, 1);
    // batch-load the whole A tile, then issue all wmmas back-to-back
    bf16x16 af[11];
#pragma unroll
    for (int kk = 0; kk < 11; ++kk) af[kk] = load_frag(arow + kk * 32, hi);
    f32x8 acc = {};
#pragma unroll
    for (int kk = 0; kk < 11; ++kk)
      acc = __builtin_amdgcn_wmma_f32_16x16x32_bf16(
          false, af[kk], false, bfrag[kk], (short)0, acc, false, false);
    SCHED_GROUP(0x0020, 23);   // all VMEM reads of this tile first
    SCHED_GROUP(0x0008, 11);   // then the WMMA burst
#pragma unroll
    for (int r = 0; r < 8; ++r) {
      int h = h0 + r + 8 * hi;           // C layout: VGPR r -> row M=r+8*hi
      float v = acc[r];
      if (h < H_ && v > bestv) { bestv = v; besth = h; }
    }
  }

  // combine the two half-lane row groups (lane <-> lane^16)
  {
    float ov = __shfl_xor(bestv, 16, 32);
    int   oh = __shfl_xor(besth, 16, 32);
    if (ov > bestv || (ov == bestv && oh < besth)) { bestv = ov; besth = oh; }
  }
  __shared__ float sv[8][16];
  __shared__ int   sh[8][16];
  if (hi == 0) { sv[wv][n] = bestv; sh[wv][n] = besth; }
  __syncthreads();
  if (wv == 0 && hi == 0) {
    float bv = sv[0][n];
    int   bh = sh[0][n];
#pragma unroll
    for (int w = 1; w < 8; ++w) {
      float v = sv[w][n];
      int   h = sh[w][n];
      if (v > bv || (v == bv && h < bh)) { bv = v; bh = h; }
    }
    if (t < T_) pos[b * T_ + t] = bh;
  }
}

// Causal 3-tap pool of argmax positions, gather HL row, threshold -> gate
__global__ void gate_kernel(const int* __restrict__ pos, const float* __restrict__ HL,
                            float* __restrict__ gate) {
  int idx = blockIdx.x * blockDim.x + threadIdx.x;   // over B*T
  if (idx >= B_ * T_) return;
  int t = idx % T_;
  int b = idx / T_;
  float p0 = (float)pos[b * T_ + t];
  float p1 = (t >= 1) ? (float)pos[b * T_ + t - 1] : 1e-8f;
  float p2 = (t >= 2) ? (float)pos[b * T_ + t - 2] : 1e-8f;
  float sm = (p2 + p1 + p0) / 3.0f;
  int hidx = (int)sm;
  if (hidx < 0) hidx = 0;
  if (hidx >= H_) hidx = H_ - 1;
  const float* row = HL + (size_t)hidx * F_;
  for (int f = 0; f < F_; ++f)
    gate[((size_t)b * F_ + f) * T_ + t] = (row[f] > 0.0f) ? 1.0f : 0.0f;
}

// ---------------------------------------------------------------------------
// GatedConv gate branch (pointwise): BN -> 1x1 -> PReLU -> 1x1 -> BN -> sigmoid,
// then writes gated activation (x * alpha) in NHWC-padded bf16.
// ---------------------------------------------------------------------------
__global__ void gatebranch_kernel(
    const bf16_t* __restrict__ act, const float* __restrict__ gate,
    bf16_t* __restrict__ gated,
    const float* __restrict__ bn1g, const float* __restrict__ bn1b,
    const float* __restrict__ bn1m, const float* __restrict__ bn1v,
    const float* __restrict__ w1, const float* __restrict__ b1,
    const float* __restrict__ p1, const float* __restrict__ w2,
    const float* __restrict__ b2,
    const float* __restrict__ bn2g, const float* __restrict__ bn2b,
    const float* __restrict__ bn2m, const float* __restrict__ bn2v,
    int cin) {
  int idx = blockIdx.x * blockDim.x + threadIdx.x;   // over B*F*T
  if (idx >= B_ * F_ * T_) return;
  int t = idx % T_;
  int f = (idx / T_) % F_;
  int b = idx / (F_ * T_);
  const int g = cin + 1;
  size_t pix = (((size_t)b * FP + (f + 2)) * TP + (t + 1)) * CP;

  float x[16];
  bf16x8 v0 = *(const bf16x8*)(act + pix);
  bf16x8 v1 = *(const bf16x8*)(act + pix + 8);
#pragma unroll
  for (int c = 0; c < 8; ++c) { x[c] = (float)v0[c]; x[c + 8] = (float)v1[c]; }
  float gg = gate[idx];

  float a[17];
#pragma unroll
  for (int c = 0; c < 17; ++c) {
    if (c < g) {
      float val = (c < cin) ? x[c] : gg;
      a[c] = (val - bn1m[c]) * bn1g[c] * rsqrtf(bn1v[c] + 1e-5f) + bn1b[c];
    }
  }
  float p1v = p1[0];
  float h[17];
#pragma unroll
  for (int c = 0; c < 17; ++c) {
    if (c < g) {
      float s = b1[c];
#pragma unroll
      for (int j = 0; j < 17; ++j)
        if (j < g) s += w1[c * g + j] * a[j];
      h[c] = (s >= 0.0f) ? s : p1v * s;
    }
  }
  float z = b2[0];
#pragma unroll
  for (int j = 0; j < 17; ++j)
    if (j < g) z += w2[j] * h[j];
  z = (z - bn2m[0]) * bn2g[0] * rsqrtf(bn2v[0] + 1e-5f) + bn2b[0];
  float alpha = 1.0f / (1.0f + __expf(-z));

  bf16x8 o0, o1;
#pragma unroll
  for (int c = 0; c < 8; ++c) {
    float q0 = (c < cin) ? x[c] * alpha : 0.0f;
    float q1 = (c + 8 < cin) ? x[c + 8] * alpha : 0.0f;
    o0[c] = (bf16_t)q0;
    o1[c] = (bf16_t)q1;
  }
  *(bf16x8*)(gated + pix)     = o0;
  *(bf16x8*)(gated + pix + 8) = o1;
}

// ---------------------------------------------------------------------------
// Causal 5x2 conv via 10 per-tap WMMA GEMMs on NHWC-padded bf16.
// Grid: (16 tile groups, 321 f, 8 b); block 32x4 (4 waves, one t-tile each).
// sched_group_barrier keeps the 35 fragment loads (20 weight + 15 activation
// chunks) ahead of the 10-wmma burst; launch_bounds relaxes the VGPR cap so
// all fragments stay resident. Per kh the kw=0/kw=1 fragments share their
// middle 16B chunk (3 loads -> 2 frags).
// Modes: 0 = store, 1 = prelu, 2 = +res then prelu, 3 = +res, BN, x gate -> f32 out.
// ---------------------------------------------------------------------------
__global__ void __launch_bounds__(128, 1)
conv5x2_kernel(
    const bf16_t* __restrict__ in, const bf16_t* __restrict__ wp,
    bf16_t* __restrict__ out, const bf16_t* __restrict__ res,
    const float* __restrict__ prelu_a,
    const float* __restrict__ bng, const float* __restrict__ bnb,
    const float* __restrict__ bnm, const float* __restrict__ bnv,
    const float* __restrict__ gate, float* __restrict__ dout,
    int mode) {
  const int tt = blockIdx.x * 4 + threadIdx.y;
  if (tt >= NTT) return;                 // wave-uniform: EXEC stays all-ones
  const int f  = blockIdx.y;
  const int b  = blockIdx.z;
  const int lane = threadIdx.x;
  const int n  = lane & 15;
  const int hi = lane >> 4;
  const int t0 = tt * 16;

  bf16x16 wf[10];
#pragma unroll
  for (int tap = 0; tap < 10; ++tap)
    wf[tap] = load_frag(wp + tap * 512 + n * 32, hi);

  bf16x16 bfr[10];
#pragma unroll
  for (int kh = 0; kh < 5; ++kh) {
    // buffer row = 2+(f+kh-2) = f+kh ; col for kw is t0+n+kw
    const bf16_t* p = in + (((size_t)b * FP + (f + kh)) * TP + (t0 + n)) * CP;
    bf16x8 c0 = *(const bf16x8*)(p + hi * 8);
    bf16x8 c1 = *(const bf16x8*)(p + 16 + hi * 8);
    bf16x8 c2 = *(const bf16x8*)(p + 32 + hi * 8);
    bfr[kh * 2 + 0] = cat8(c0, c1);
    bfr[kh * 2 + 1] = cat8(c1, c2);
  }

  f32x8 acc = {};
#pragma unroll
  for (int tap = 0; tap < 10; ++tap)
    acc = __builtin_amdgcn_wmma_f32_16x16x32_bf16(
        false, wf[tap], false, bfr[tap], (short)0, acc, false, false);
  SCHED_GROUP(0x0020, 35);   // all fragment loads first
  SCHED_GROUP(0x0008, 10);   // then the WMMA burst

  const int t = t0 + n;
  size_t opix = (((size_t)b * FP + (f + 2)) * TP + (t + 1)) * CP;

  if (mode == 3) {  // final layer: cout=1 lives in row M=0 (hi==0, r==0)
    if (hi == 0) {
      bf16x8 rv = *(const bf16x8*)(res + opix);
      float v = acc[0] + (float)rv[0];
      v = (v - bnm[0]) * bng[0] * rsqrtf(bnv[0] + 1e-5f) + bnb[0];
      if (t < T_) {
        size_t oi = ((size_t)b * F_ + f) * T_ + t;
        dout[oi] = v * gate[oi];
      }
    }
    return;
  }

  float av = (mode >= 1) ? prelu_a[0] : 0.0f;
  bf16x8 rv = {};
  if (mode == 2) rv = *(const bf16x8*)(res + opix + hi * 8);
  bf16x8 o;
#pragma unroll
  for (int r = 0; r < 8; ++r) {
    float v = acc[r];
    if (mode == 2) v += (float)rv[r];
    if (mode >= 1) v = (v >= 0.0f) ? v : av * v;
    o[r] = (bf16_t)v;
  }
  *(bf16x8*)(out + opix + hi * 8) = o;   // lane writes 8 consecutive channels
}

// ---------------------------------------------------------------------------
// Host launch
// ---------------------------------------------------------------------------
#define PARAM(k) ((const float*)d_in[(k)])

extern "C" void kernel_launch(void* const* d_in, const int* in_sizes, int n_in,
                              void* d_out, int out_size, void* d_ws, size_t ws_size,
                              hipStream_t stream) {
  (void)in_sizes; (void)n_in; (void)out_size; (void)ws_size;
  char* ws = (char*)d_ws;
  bf16_t* magT = (bf16_t*)(ws + OFF_MAGT);
  bf16_t* Mbf  = (bf16_t*)(ws + OFF_MBF);
  int*    pos  = (int*)   (ws + OFF_POS);
  float*  gate = (float*) (ws + OFF_GATE);
  bf16_t* wpk  = (bf16_t*)(ws + OFF_WPK);
  bf16_t* buf0 = (bf16_t*)(ws + OFF_BUF0);
  bf16_t* buf1 = (bf16_t*)(ws + OFF_BUF1);
  bf16_t* buf2 = (bf16_t*)(ws + OFF_BUF2);

  const float* mag = (const float*)d_in[0];
  const float* inf = (const float*)d_in[1];
  const float* M   = (const float*)d_in[2];
  const float* HL  = (const float*)d_in[3];

  const int NBFT = B_ * F_ * T_;
  const int GBFT = (NBFT + 255) / 256;

  // 1. zero entire workspace (padding invariants depend on it)
  zero_kernel<<<4096, 256, 0, stream>>>((uint32_t*)ws, WS_TOTAL / 4);

  // 2. bf16 conversions / packing
  cvt_magT_kernel<<<GBFT, 256, 0, stream>>>(mag, magT);
  cvt_M_kernel<<<(H_ * F_ + 255) / 256, 256, 0, stream>>>(M, Mbf);
  init_act_kernel<<<GBFT, 256, 0, stream>>>(inf, buf0);

  static const int Lbase[4] = {4, 22, 40, 58};
  static const int CIN[4]  = {1, 8, 16, 8};
  static const int COUT[4] = {8, 16, 8, 1};
  for (int i = 0; i < 4; ++i) {
    pack_w_kernel<<<20, 256, 0, stream>>>(PARAM(Lbase[i] + 13),
                                          wpk + (size_t)(i * 3 + 0) * 5120, COUT[i], CIN[i]);
    pack_w_kernel<<<20, 256, 0, stream>>>(PARAM(Lbase[i] + 14),
                                          wpk + (size_t)(i * 3 + 1) * 5120, COUT[i], COUT[i]);
    pack_w_kernel<<<20, 256, 0, stream>>>(PARAM(Lbase[i] + 16),
                                          wpk + (size_t)(i * 3 + 2) * 5120, COUT[i], COUT[i]);
  }

  // 3. harmonic integral: WMMA GEMM + argmax, then pool/gather/threshold
  {
    dim3 g(NTT, B_);
    dim3 blk(32, 8);
    gemm_argmax_kernel<<<g, blk, 0, stream>>>(magT, Mbf, pos);
    gate_kernel<<<(B_ * T_ + 255) / 256, 256, 0, stream>>>(pos, HL, gate);
  }

  // 4. GHCM: 4 gated layers
  dim3 cg((NTT + 3) / 4, F_, B_);
  dim3 cb(32, 4);
  for (int i = 0; i < 4; ++i) {
    const int L = Lbase[i];
    gatebranch_kernel<<<GBFT, 256, 0, stream>>>(
        buf0, gate, buf1,
        PARAM(L + 0), PARAM(L + 1), PARAM(L + 2), PARAM(L + 3),
        PARAM(L + 4), PARAM(L + 5), PARAM(L + 6), PARAM(L + 7), PARAM(L + 8),
        PARAM(L + 9), PARAM(L + 10), PARAM(L + 11), PARAM(L + 12), CIN[i]);
    // out_lin = conv(gated)                      buf1 -> buf2
    conv5x2_kernel<<<cg, cb, 0, stream>>>(
        buf1, wpk + (size_t)(i * 3 + 0) * 5120, buf2, nullptr,
        nullptr, nullptr, nullptr, nullptr, nullptr, nullptr, nullptr, 0);
    // r1 = prelu(conv(out_lin))                  buf2 -> buf1
    conv5x2_kernel<<<cg, cb, 0, stream>>>(
        buf2, wpk + (size_t)(i * 3 + 1) * 5120, buf1, nullptr,
        PARAM(L + 15), nullptr, nullptr, nullptr, nullptr, nullptr, nullptr, 1);
    if (i < 3) {
      // act = prelu(conv(r1) + out_lin)          buf1 (+buf2) -> buf0
      conv5x2_kernel<<<cg, cb, 0, stream>>>(
          buf1, wpk + (size_t)(i * 3 + 2) * 5120, buf0, buf2,
          PARAM(L + 17), nullptr, nullptr, nullptr, nullptr, nullptr, nullptr, 2);
    } else {
      // d_out = bn(conv(r1) + out_lin) * gate
      conv5x2_kernel<<<cg, cb, 0, stream>>>(
          buf1, wpk + (size_t)(i * 3 + 2) * 5120, nullptr, buf2,
          nullptr, PARAM(L + 17), PARAM(L + 18), PARAM(L + 19), PARAM(L + 20),
          gate, (float*)d_out, 3);
    }
  }
}
